// DoubleSubstitutionEmbedding_7791070675697
// MI455X (gfx1250) — matrix operand order repaired and checked by reading, not compiled, not run on hardware
//
#include <hip/hip_runtime.h>
#include <hip/hip_bf16.h>

// Problem constants (from the reference setup)
#define B_    2
#define L2_   1024
#define L1_   8192
#define L0_   65536
#define S_    (L2_ + L1_ + L0_)   // 74752
#define D_    256
#define RES_  32

typedef __attribute__((ext_vector_type(16))) __bf16 v16bf;
typedef __attribute__((ext_vector_type(8)))  __bf16 v8bf;
typedef __attribute__((ext_vector_type(8)))  float  v8f;

// ---------------------------------------------------------------------------
// Embedding: e = emb_val[v] + emb_dep[dep] + sum_i emb_pos[i][pos_i], -> bf16
// 4 sequence positions per 256-thread block.
// ---------------------------------------------------------------------------
__global__ __launch_bounds__(256) void embed_kernel(
    const int* __restrict__ value, const int* __restrict__ depth,
    const int* __restrict__ position,
    const float* __restrict__ emb_val, const float* __restrict__ emb_dep,
    const float* __restrict__ emb_pos, __bf16* __restrict__ X)
{
  const int b = blockIdx.y, d = threadIdx.x;
  const int s0 = blockIdx.x * 4;
#pragma unroll
  for (int e = 0; e < 4; ++e) {
    const long long bs = (long long)b * S_ + s0 + e;
    const int v = value[bs], dep = depth[bs];
    const int* pos = position + bs * 3;
    float acc = emb_val[v * D_ + d] + emb_dep[dep * D_ + d];
#pragma unroll
    for (int i = 0; i < 3; ++i)
      acc += emb_pos[(i * (RES_ + 1) + pos[i]) * D_ + d];
    X[bs * D_ + d] = (__bf16)acc;
  }
}

// ---------------------------------------------------------------------------
// Weight prep: W[o, d, k] (fp32) -> bf16 B-fragment layout.
// Fragment flat index f = ((kt*16 + t)*32 + lane)*16 + j, where
//   lane = half*16 + n_local ; element j -> K = kt*32 + half*8 + j (j<8), kt*32 + 16 + half*8 + (j-8) (j>=8),
//   N = t*16 + n_local.  Contraction index kk = k_tap*256 + d_channel
// (matches the row-major (l,8,D) A view).
// ---------------------------------------------------------------------------
__global__ __launch_bounds__(256) void prep_w_kernel(
    const float* __restrict__ W, __bf16* __restrict__ Bf, int K, int s)
{
  const int f = blockIdx.x * 256 + threadIdx.x;
  if (f >= K * D_) return;
  const int j    = f & 15;
  const int lane = (f >> 4) & 31;
  const int t    = (f >> 9) & 15;
  const int kt   = f >> 13;
  const int half = lane >> 4, nl = lane & 15;
  const int kk = kt * 32 + (j < 8 ? half * 8 + j : 16 + half * 8 + (j - 8));
  const int n  = t * 16 + nl;
  const int ktap = kk >> 8, d = kk & 255;
  Bf[f] = (__bf16)W[((long long)n * D_ + d) * s + ktap];
}

// ---------------------------------------------------------------------------
// GEMM  Y[M,256] = A[M,K](bf16) x Bfrag + bias.
// Block = 8 waves; wave w owns rows [blockIdx.x*128 + 16w, +16).
// blockIdx.z selects 8 of the 16 N-tiles (64 acc VGPRs per wave).
// Software-pipelined: B fragments are double-buffered in registers (bA/bB)
// so each 4-wmma group consumes loads issued one group earlier.
// ---------------------------------------------------------------------------
template <typename OutT>
__global__ __launch_bounds__(256) void gemm_bias_kernel(
    const __bf16* __restrict__ A, long long strideAb,
    const __bf16* __restrict__ Bf,
    const float* __restrict__ bias,
    OutT* __restrict__ Y, long long strideYb,
    int K)
{
  const int tid  = threadIdx.x;
  const int wave = tid >> 5, lane = tid & 31;
  const int m0   = blockIdx.x * 128 + wave * 16;
  const int t0   = blockIdx.z * 8;
  const __bf16* Ab = A + (long long)blockIdx.y * strideAb;
  OutT*         Yb = Y + (long long)blockIdx.y * strideYb;

  // A fragment addressing per ISA 16x32 bf16 A layout:
  // lane<16: K {0..7,16..23}; lane>=16: K {8..15,24..31}, row M = lane&15.
  const int half = lane >> 4, mrow = lane & 15;
  const __bf16* arow  = Ab + (long long)(m0 + mrow) * K + half * 8;
  const __bf16* bbase = Bf + (size_t)t0 * 512 + (size_t)lane * 16;

  v8f acc[8] = {};
  const int nkt = K >> 5;

  // Prologue: A frag + first 4 B frags for kt=0.
  v8bf lo = *(const v8bf*)(arow);
  v8bf hi = *(const v8bf*)(arow + 16);
  v16bf a_cur = __builtin_shufflevector(lo, hi,
      0, 1, 2, 3, 4, 5, 6, 7, 8, 9, 10, 11, 12, 13, 14, 15);
  v16bf bA[4], bB[4];
#pragma unroll
  for (int t = 0; t < 4; ++t)
    bA[t] = *(const v16bf*)(bbase + (size_t)t * 512);

  for (int kt = 0; kt < nkt; ++kt) {
    const int ktn = (kt + 1 < nkt) ? (kt + 1) : kt;   // clamped, branch-free
    const __bf16* bk  = bbase + (size_t)kt  * 8192;
    const __bf16* bkn = bbase + (size_t)ktn * 8192;

    // Issue loads for the second tile group + next A fragment.
#pragma unroll
    for (int t = 0; t < 4; ++t)
      bB[t] = *(const v16bf*)(bk + (size_t)(4 + t) * 512);
    v8bf lo2 = *(const v8bf*)(arow + (size_t)ktn * 32);
    v8bf hi2 = *(const v8bf*)(arow + (size_t)ktn * 32 + 16);
    v16bf a_nxt = __builtin_shufflevector(lo2, hi2,
        0, 1, 2, 3, 4, 5, 6, 7, 8, 9, 10, 11, 12, 13, 14, 15);

    // Tile group 0: operands loaded one group ago (latency already covered).
#pragma unroll
    for (int t = 0; t < 4; ++t)
      acc[t] = __builtin_amdgcn_wmma_f32_16x16x32_bf16(
          false, a_cur, false, bA[t], (short)0, acc[t], false, false);

    // Refill bA for the next iteration while group-1 wmma runs.
#pragma unroll
    for (int t = 0; t < 4; ++t)
      bA[t] = *(const v16bf*)(bkn + (size_t)t * 512);

    // Tile group 1: its loads overlapped the group-0 wmma chain.
#pragma unroll
    for (int t = 0; t < 4; ++t)
      acc[4 + t] = __builtin_amdgcn_wmma_f32_16x16x32_bf16(
          false, a_cur, false, bB[t], (short)0, acc[4 + t], false, false);

    a_cur = a_nxt;
  }

  // C/D layout: VGPR r, lanes 0-15 -> M=r, lanes 16-31 -> M=8+r, N=lane&15.
  const int col = lane & 15, rh = lane >> 4;
#pragma unroll
  for (int t = 0; t < 8; ++t) {
    const int n = (t0 + t) * 16 + col;
    const float bv = bias[n];
#pragma unroll
    for (int r = 0; r < 8; ++r)
      Yb[(size_t)(m0 + rh * 8 + r) * D_ + n] = (OutT)(acc[t][r] + bv);
  }
}

// ---------------------------------------------------------------------------
// Block-wide exclusive scan, 1024 threads, LDS Hillis-Steele.
// ---------------------------------------------------------------------------
__device__ __forceinline__ int block_scan_excl(int x, int* tmp, int* total)
{
  const int tid = threadIdx.x;
  tmp[tid] = x;
  __syncthreads();
  for (int off = 1; off < 1024; off <<= 1) {
    int v = (tid >= off) ? tmp[tid - off] : 0;
    __syncthreads();
    tmp[tid] += v;
    __syncthreads();
  }
  const int incl = tmp[tid];
  const int tot  = tmp[1023];
  __syncthreads();
  *total = tot;
  return incl - x;
}

// ---------------------------------------------------------------------------
// Substitution index: general _substitute() index computation per batch.
//   mask_src[j] = value[src_off + 8j] != 0 ; stable partition -> order[]
//   mask_sub[l] = value[tgt_off + l] == 2 ; idx = clip(cumsum-1, 0, L-1)
//   g[l] = mask_sub ? order[idx] : -1
// One block per batch, 1024 threads, E = L/1024 elems/thread.
// ---------------------------------------------------------------------------
__global__ __launch_bounds__(1024) void build_sub_index_kernel(
    const int* __restrict__ value, int tgt_off, int src_off, int L,
    int* __restrict__ g_out)
{
  __shared__ int order_s[8192];
  __shared__ int tmp_s[1024];
  const int b = blockIdx.x, tid = threadIdx.x;
  const int E = L >> 10;
  const long long vb = (long long)b * S_;

  int loc[8];
  int cnt = 0;
  for (int e = 0; e < E; ++e) {
    const int j = tid * E + e;
    const int ms = value[vb + src_off + 8 * j] != 0;
    loc[e] = ms; cnt += ms;
  }
  int total;
  int run = block_scan_excl(cnt, tmp_s, &total);
  for (int e = 0; e < E; ++e) {
    const int j = tid * E + e;
    if (loc[e]) { order_s[run] = j; run++; }
    else        { order_s[total + (j - run)] = j; }
  }
  __syncthreads();

  cnt = 0;
  for (int e = 0; e < E; ++e) {
    const int l = tid * E + e;
    const int ms = value[vb + tgt_off + l] == 2;
    loc[e] = ms; cnt += ms;
  }
  int tot2;
  run = block_scan_excl(cnt, tmp_s, &tot2);
  for (int e = 0; e < E; ++e) {
    const int l = tid * E + e;
    int gi = -1;
    if (loc[e]) {
      int c = run; run++;
      if (c > L - 1) c = L - 1;
      gi = order_s[c];
    }
    g_out[(long long)b * L + l] = gi;
  }
}

// ---------------------------------------------------------------------------
// Apply substitution: A_next[l,:] = g>=0 ? y_bf16[g,:] : X_emb[l,:]
// 4 rows per block.
// ---------------------------------------------------------------------------
__global__ __launch_bounds__(256) void subst_kernel(
    const int* __restrict__ g, const __bf16* __restrict__ y,
    const __bf16* __restrict__ X, int emb_off, int L,
    __bf16* __restrict__ Aout)
{
  const int b = blockIdx.y, d = threadIdx.x;
  const int l0 = blockIdx.x * 4;
#pragma unroll
  for (int e = 0; e < 4; ++e) {
    const int l = l0 + e;
    const int gi = g[(long long)b * L + l];
    __bf16 o;
    if (gi >= 0) o = y[((size_t)b * L + gi) * D_ + d];
    else         o = X[((size_t)b * S_ + emb_off + l) * D_ + d];
    Aout[((size_t)b * L + l) * D_ + d] = o;
  }
}

// ---------------------------------------------------------------------------
// Final mask / output compaction index (m1 -> m2 -> m2g -> order2, nvalid)
// ---------------------------------------------------------------------------
__global__ __launch_bounds__(1024) void final_mask_kernel(
    const int* __restrict__ value, int* __restrict__ order2,
    int* __restrict__ nvalid)
{
  __shared__ int m1s[1024];
  __shared__ int m2s[1024];
  __shared__ int tmp_s[1024];
  const int b = blockIdx.x, tid = threadIdx.x;
  const long long vb = (long long)b * S_;

  int m1 = 0;
#pragma unroll
  for (int k = 0; k < 8; ++k) m1 |= (value[vb + L2_ + 8 * tid + k] == 2);
  m1s[tid] = m1;

  const int is2 = value[vb + tid] == 2;
  int tot;
  const int excl = block_scan_excl(is2, tmp_s, &tot);
  int idx = excl + is2 - 1;
  if (idx < 0) idx = 0;
  if (idx > 1023) idx = 1023;
  const int m2 = is2 ? m1s[idx] : 0;
  m2s[tid] = m2;
  __syncthreads();

  int m2g = 0;
  if (tid < 256) {
#pragma unroll
    for (int k = 0; k < 4; ++k) m2g |= m2s[4 * tid + k];
  }
  const int x = (tid < 256) ? m2g : 0;
  int totg;
  const int e2 = block_scan_excl(x, tmp_s, &totg);
  if (tid < 256) {
    const int pos = m2g ? e2 : (totg + (tid - e2));
    order2[b * 256 + pos] = tid;
  }
  if (tid == 0) nvalid[b] = totg;
}

__global__ __launch_bounds__(256) void gather_out_kernel(
    const float* __restrict__ xout, const int* __restrict__ order2,
    const int* __restrict__ nvalid, float* __restrict__ out)
{
  const int p = blockIdx.x, b = blockIdx.y, n = threadIdx.x;
  const int src = order2[b * 256 + p];
  const float v = (p < nvalid[b]) ? xout[((size_t)b * 256 + src) * D_ + n] : 0.f;
  out[((size_t)b * 256 + p) * D_ + n] = v;
}

// ---------------------------------------------------------------------------
extern "C" void kernel_launch(void* const* d_in, const int* in_sizes, int n_in,
                              void* d_out, int out_size, void* d_ws, size_t ws_size,
                              hipStream_t stream)
{
  (void)in_sizes; (void)n_in; (void)out_size; (void)ws_size;
  const int*   value    = (const int*)d_in[0];
  const int*   depth    = (const int*)d_in[1];
  const int*   position = (const int*)d_in[2];
  const float* emb_val  = (const float*)d_in[3];
  const float* emb_dep  = (const float*)d_in[4];
  const float* emb_pos  = (const float*)d_in[5];
  const float* W0 = (const float*)d_in[6];
  const float* b0 = (const float*)d_in[7];
  const float* W1 = (const float*)d_in[8];
  const float* b1 = (const float*)d_in[9];
  const float* W2 = (const float*)d_in[10];
  const float* b2 = (const float*)d_in[11];
  float* out = (float*)d_out;

  char* ws = (char*)d_ws;
  auto alloc = [&](size_t bytes) -> char* {
    char* p = ws;
    ws += (bytes + 255) & ~(size_t)255;
    return p;
  };
  __bf16* X    = (__bf16*)alloc((size_t)B_ * S_ * D_ * 2);   // 76.5 MB
  __bf16* Bf0  = (__bf16*)alloc((size_t)2048 * D_ * 2);
  __bf16* Bf1  = (__bf16*)alloc((size_t)2048 * D_ * 2);
  __bf16* Bf2  = (__bf16*)alloc((size_t)1024 * D_ * 2);
  __bf16* y0   = (__bf16*)alloc((size_t)B_ * L1_ * D_ * 2);
  __bf16* A1   = (__bf16*)alloc((size_t)B_ * L1_ * D_ * 2);
  __bf16* y1   = (__bf16*)alloc((size_t)B_ * L2_ * D_ * 2);
  __bf16* A2   = (__bf16*)alloc((size_t)B_ * L2_ * D_ * 2);
  float*  xo   = (float*) alloc((size_t)B_ * 256 * D_ * 4);
  int* g1      = (int*)   alloc((size_t)B_ * L1_ * 4);
  int* g2      = (int*)   alloc((size_t)B_ * L2_ * 4);
  int* order2  = (int*)   alloc((size_t)B_ * 256 * 4);
  int* nvalid  = (int*)   alloc((size_t)B_ * 4);

  // Weight prep + embeddings + index computations (independent)
  prep_w_kernel<<<2048, 256, 0, stream>>>(W0, Bf0, 2048, 8);
  prep_w_kernel<<<2048, 256, 0, stream>>>(W1, Bf1, 2048, 8);
  prep_w_kernel<<<1024, 256, 0, stream>>>(W2, Bf2, 1024, 4);
  embed_kernel<<<dim3(S_ / 4, B_), 256, 0, stream>>>(
      value, depth, position, emb_val, emb_dep, emb_pos, X);
  build_sub_index_kernel<<<B_, 1024, 0, stream>>>(value, L2_, L2_ + L1_, L1_, g1);
  build_sub_index_kernel<<<B_, 1024, 0, stream>>>(value, 0, L2_, L2_, g2);
  final_mask_kernel<<<B_, 1024, 0, stream>>>(value, order2, nvalid);

  // conv0: A = X[:, L2+L1:] viewed as (8192 x 2048) per batch -> y0 (bf16)
  gemm_bias_kernel<__bf16><<<dim3(8192 / 128, B_, 2), 256, 0, stream>>>(
      X + (size_t)(L2_ + L1_) * D_, (long long)S_ * D_,
      Bf0, b0, y0, (long long)L1_ * D_, 2048);
  subst_kernel<<<dim3(L1_ / 4, B_), 256, 0, stream>>>(g1, y0, X, L2_, L1_, A1);

  // conv1: A1 viewed as (1024 x 2048) per batch -> y1 (bf16)
  gemm_bias_kernel<__bf16><<<dim3(1024 / 128, B_, 2), 256, 0, stream>>>(
      A1, (long long)L1_ * D_, Bf1, b1, y1, (long long)L2_ * D_, 2048);
  subst_kernel<<<dim3(L2_ / 4, B_), 256, 0, stream>>>(g2, y1, X, 0, L2_, A2);

  // conv2: A2 viewed as (256 x 1024) per batch -> xo (fp32)
  gemm_bias_kernel<float><<<dim3(256 / 128, B_, 2), 256, 0, stream>>>(
      A2, (long long)L2_ * D_, Bf2, b2, xo, (long long)256 * D_, 1024);

  gather_out_kernel<<<dim3(256, B_), 256, 0, stream>>>(xo, order2, nvalid, out);
}
